// NNCLR_27462020891212
// MI455X (gfx1250) — matrix operand order repaired and checked by reading, not compiled, not run on hardware
//
#include <hip/hip_runtime.h>
#include <hip/hip_bf16.h>

typedef __bf16 bf16;
typedef __attribute__((ext_vector_type(16))) __bf16 v16bf;
typedef __attribute__((ext_vector_type(8)))  float  v8f;
typedef __attribute__((ext_vector_type(4)))  unsigned int u32x4;
typedef unsigned int uint32;

#define WG 256
#define LDS_STRIDE 40   // bf16 elements per LDS row (80B: 16B-aligned, bank-staggered)

// ---------------------------------------------------------------------------
// Utility kernels
// ---------------------------------------------------------------------------
__global__ void k_f32_to_bf16(const float* __restrict__ src, bf16* __restrict__ dst, long n) {
    long i = (long)blockIdx.x * blockDim.x + threadIdx.x;
    long stride = (long)gridDim.x * blockDim.x;
    for (; i < n; i += stride) dst[i] = (bf16)src[i];
}

__global__ void k_copy_f32(const float* __restrict__ src, float* __restrict__ dst, long n) {
    long i = (long)blockIdx.x * blockDim.x + threadIdx.x;
    long stride = (long)gridDim.x * blockDim.x;
    for (; i < n; i += stride) dst[i] = src[i];
}

__global__ void k_zero_f32(float* __restrict__ dst, long n) {
    long i = (long)blockIdx.x * blockDim.x + threadIdx.x;
    long stride = (long)gridDim.x * blockDim.x;
    for (; i < n; i += stride) dst[i] = 0.0f;
}

// ---------------------------------------------------------------------------
// Double-buffered WMMA GEMM main loop.
// C[M x N] = A[M x K] * B  (bt==0: B is K x N row-major; bt==1: B is N x K)
// Block tile 128x64, 8 wave32 waves, each wave: 2x2 tiles of 16x16x32 bf16.
// Global loads are b128 (8 bf16); LDS staged double-buffered so the next
// tile's global loads are in flight while the current tile's WMMAs execute.
// ---------------------------------------------------------------------------
__device__ __forceinline__ void wmma_mainloop(
    const bf16* __restrict__ A, const bf16* __restrict__ B, int bt,
    int N, int K, int bm, int bn,
    bf16* sA, bf16* sB, v8f acc[2][2])
{
    const int tid  = threadIdx.x;
    const int lane = tid & 31;
    const int wave = tid >> 5;
    const int wm   = wave >> 1;     // 0..3  (M direction, 32 rows each)
    const int wn   = wave & 1;      // 0..1  (N direction, 32 cols each)
    const int l    = lane & 15;
    const int hk   = lane >> 4;     // lane half selects K sub-block

    const u32x4* Aq = (const u32x4*)A;   // 8 bf16 per u32x4
    const u32x4* Bq = (const u32x4*)B;
    const int Kq = K >> 3;
    const int Nq = N >> 3;

    // per-thread staging indices
    int ar[2], akq[2];
    #pragma unroll
    for (int i = 0; i < 2; ++i) {
        int idx = tid + i * WG;
        ar[i]  = idx >> 2;          // 0..127
        akq[i] = idx & 3;           // 0..3 (x8 bf16)
    }
    const int br  = tid >> 2;       // bt path: 0..63
    const int bkq = tid & 3;
    const int bk  = tid >> 3;       // !bt path: 0..31 (k row)
    const int bnq = tid & 7;        // 0..7  (x8 n values)

    u32x4 ra[2], rb;
    auto gload = [&](int kt) {
        #pragma unroll
        for (int i = 0; i < 2; ++i)
            ra[i] = Aq[(long)(bm + ar[i]) * Kq + (kt >> 3) + akq[i]];
        if (bt) rb = Bq[(long)(bn + br) * Kq + (kt >> 3) + bkq];
        else    rb = Bq[(long)(kt + bk) * Nq + (bn >> 3) + bnq];
    };
    auto sstore = [&](int buf) {
        bf16* dA = sA + buf * (128 * LDS_STRIDE);
        bf16* dB = sB + buf * (64 * LDS_STRIDE);
        #pragma unroll
        for (int i = 0; i < 2; ++i)
            *(u32x4*)(dA + ar[i] * LDS_STRIDE + akq[i] * 8) = ra[i];
        if (bt) {
            *(u32x4*)(dB + br * LDS_STRIDE + bkq * 8) = rb;
        } else {
            union { u32x4 v; bf16 h[8]; } cv; cv.v = rb;
            #pragma unroll
            for (int e = 0; e < 8; ++e)
                dB[(bnq * 8 + e) * LDS_STRIDE + bk] = cv.h[e];
        }
    };

    // prologue: stage tile 0
    gload(0);
    sstore(0);
    __syncthreads();

    int buf = 0;
    for (int kt = 0; kt < K; kt += 32, buf ^= 1) {
        const bool hasNext = (kt + 32) < K;
        if (hasNext) gload(kt + 32);               // loads in flight during WMMA
        if (kt + 64 < K)                            // hint tile after next into L2
            __builtin_prefetch(&Aq[(long)(bm + ar[0]) * Kq + ((kt + 64) >> 3)], 0, 1);

        const bf16* cA = sA + buf * (128 * LDS_STRIDE);
        const bf16* cB = sB + buf * (64 * LDS_STRIDE);

        // Fragments in contiguous k order (k = 16*hk + s), same for A and B
        v16bf af[2], bfr[2];
        #pragma unroll
        for (int mi = 0; mi < 2; ++mi) {
            const bf16* ap = cA + (wm * 32 + mi * 16 + l) * LDS_STRIDE;
            const int kb = hk * 16;
            #pragma unroll
            for (int j = 0; j < 8; ++j) af[mi][j] = ap[kb + j];
            #pragma unroll
            for (int j = 0; j < 8; ++j) af[mi][8 + j] = ap[kb + 8 + j];
        }
        #pragma unroll
        for (int ni = 0; ni < 2; ++ni) {
            const bf16* bp = cB + (wn * 32 + ni * 16 + l) * LDS_STRIDE;
            const int kb = hk * 16;
            #pragma unroll
            for (int j = 0; j < 16; ++j) bfr[ni][j] = bp[kb + j];
        }
        #pragma unroll
        for (int mi = 0; mi < 2; ++mi)
            #pragma unroll
            for (int ni = 0; ni < 2; ++ni)
                acc[mi][ni] = __builtin_amdgcn_wmma_f32_16x16x32_bf16(
                    false, af[mi], false, bfr[ni], (short)0, acc[mi][ni], false, false);

        if (hasNext) sstore(buf ^ 1);              // commit next tile after compute
        __syncthreads();
    }
}

// GEMM with plain store (+ optional bias)
__global__ __launch_bounds__(WG) void k_gemm(
    const bf16* __restrict__ A, const bf16* __restrict__ B,
    float* __restrict__ C, const float* __restrict__ bias,
    int M, int N, int K, int bt)
{
    __shared__ __align__(16) bf16 sA[2 * 128 * LDS_STRIDE];
    __shared__ __align__(16) bf16 sB[2 * 64 * LDS_STRIDE];
    (void)M;
    const int bm = blockIdx.y * 128;
    const int bn = blockIdx.x * 64;
    v8f acc[2][2];
    #pragma unroll
    for (int mi = 0; mi < 2; ++mi)
        #pragma unroll
        for (int ni = 0; ni < 2; ++ni)
            #pragma unroll
            for (int r = 0; r < 8; ++r) acc[mi][ni][r] = 0.0f;

    wmma_mainloop(A, B, bt, N, K, bm, bn, sA, sB, acc);

    const int lane = threadIdx.x & 31;
    const int wave = threadIdx.x >> 5;
    const int wm = wave >> 1, wn = wave & 1;
    const int l = lane & 15, hk = lane >> 4;
    #pragma unroll
    for (int mi = 0; mi < 2; ++mi) {
        #pragma unroll
        for (int ni = 0; ni < 2; ++ni) {
            int col = bn + wn * 32 + ni * 16 + l;
            float bv = bias ? bias[col] : 0.0f;
            int rowBase = bm + wm * 32 + mi * 16 + hk * 8;
            #pragma unroll
            for (int r = 0; r < 8; ++r)
                C[(long)(rowBase + r) * N + col] = acc[mi][ni][r] + bv;
        }
    }
}

// GEMM (bt==1 path) with fused per-row max over the 64-col tile (cosine argmax)
__global__ __launch_bounds__(WG) void k_gemm_rowmax(
    const bf16* __restrict__ A, const bf16* __restrict__ Bt,
    int M, int N, int K,
    float* __restrict__ pmax, int* __restrict__ pidx)
{
    __shared__ __align__(16) bf16 sA[2 * 128 * LDS_STRIDE];
    __shared__ __align__(16) bf16 sB[2 * 64 * LDS_STRIDE];
    __shared__ float sC[128 * 66];
    (void)M;
    const int bm = blockIdx.y * 128;
    const int bn = blockIdx.x * 64;
    v8f acc[2][2];
    #pragma unroll
    for (int mi = 0; mi < 2; ++mi)
        #pragma unroll
        for (int ni = 0; ni < 2; ++ni)
            #pragma unroll
            for (int r = 0; r < 8; ++r) acc[mi][ni][r] = 0.0f;

    wmma_mainloop(A, Bt, 1, N, K, bm, bn, sA, sB, acc);

    const int lane = threadIdx.x & 31;
    const int wave = threadIdx.x >> 5;
    const int wm = wave >> 1, wn = wave & 1;
    const int l = lane & 15, hk = lane >> 4;
    #pragma unroll
    for (int mi = 0; mi < 2; ++mi) {
        #pragma unroll
        for (int ni = 0; ni < 2; ++ni) {
            int cl = wn * 32 + ni * 16 + l;
            int rb = wm * 32 + mi * 16 + hk * 8;
            #pragma unroll
            for (int r = 0; r < 8; ++r)
                sC[(rb + r) * 66 + cl] = acc[mi][ni][r];
        }
    }
    __syncthreads();
    if (threadIdx.x < 128) {
        int r = threadIdx.x;
        float best = -3.0e38f; int bc = 0;
        for (int c = 0; c < 64; ++c) {
            float v = sC[r * 66 + c];
            if (v > best) { best = v; bc = c; }
        }
        long row = bm + r;
        pmax[row * gridDim.x + blockIdx.x] = best;
        pidx[row * gridDim.x + blockIdx.x] = bn + bc;
    }
}

// ---------------------------------------------------------------------------
// BatchNorm: per-feature batch stats (one block per column, M rows)
// ---------------------------------------------------------------------------
__global__ __launch_bounds__(WG) void k_bn_stats(
    const float* __restrict__ X, int M, int N,
    float* __restrict__ mean, float* __restrict__ var)
{
    const int col = blockIdx.x;
    float s = 0.0f, s2 = 0.0f;
    for (int r = threadIdx.x; r < M; r += WG) {
        float v = X[(long)r * N + col];
        s += v; s2 += v * v;
    }
    __shared__ float rs[WG], rq[WG];
    rs[threadIdx.x] = s; rq[threadIdx.x] = s2;
    __syncthreads();
    for (int w = WG / 2; w > 0; w >>= 1) {
        if (threadIdx.x < w) {
            rs[threadIdx.x] += rs[threadIdx.x + w];
            rq[threadIdx.x] += rq[threadIdx.x + w];
        }
        __syncthreads();
    }
    if (threadIdx.x == 0) {
        float m = rs[0] / (float)M;
        mean[col] = m;
        var[col]  = rq[0] / (float)M - m * m;
    }
}

__global__ void k_bn_apply(
    const float* __restrict__ X, const float* __restrict__ mean,
    const float* __restrict__ var, const float* __restrict__ gamma,
    const float* __restrict__ beta, int M, int N, int relu,
    bf16* __restrict__ outb, float* __restrict__ outf)
{
    long n = (long)M * N;
    long i = (long)blockIdx.x * blockDim.x + threadIdx.x;
    long stride = (long)gridDim.x * blockDim.x;
    for (; i < n; i += stride) {
        int col = (int)(i % N);
        float y = gamma[col] * (X[i] - mean[col]) * rsqrtf(var[col] + 1e-5f) + beta[col];
        if (relu) y = fmaxf(y, 0.0f);
        if (outb) outb[i] = (bf16)y;
        if (outf) outf[i] = y;
    }
}

// l2-normalize rows of width 256 -> bf16
__global__ __launch_bounds__(256) void k_l2norm_rows(
    const float* __restrict__ X, bf16* __restrict__ out)
{
    const int row = blockIdx.x;
    const int t = threadIdx.x;
    float v = X[(long)row * 256 + t];
    __shared__ float rs[256];
    rs[t] = v * v;
    __syncthreads();
    for (int w = 128; w > 0; w >>= 1) { if (t < w) rs[t] += rs[t + w]; __syncthreads(); }
    float inv = 1.0f / fmaxf(sqrtf(rs[0]), 1e-12f);
    out[(long)row * 256 + t] = (bf16)(v * inv);
}

// Reduce per-tile (max, idx) partials, gather nearest-neighbour row into target
__global__ __launch_bounds__(256) void k_argmax_gather(
    const float* __restrict__ pmax, const int* __restrict__ pidx, int PN,
    const float* __restrict__ queue, const float* __restrict__ zsrc, int Q,
    float* __restrict__ target)
{
    const int row = blockIdx.x;
    const int t = threadIdx.x;
    float best = -3.0e38f; int bi = 0x7fffffff;
    for (int j = t; j < PN; j += 256) {
        float v = pmax[(long)row * PN + j];
        int ix = pidx[(long)row * PN + j];
        if (v > best || (v == best && ix < bi)) { best = v; bi = ix; }
    }
    __shared__ float bv[256]; __shared__ int bidx[256];
    bv[t] = best; bidx[t] = bi;
    __syncthreads();
    for (int w = 128; w > 0; w >>= 1) {
        if (t < w) {
            if (bv[t + w] > bv[t] || (bv[t + w] == bv[t] && bidx[t + w] < bidx[t])) {
                bv[t] = bv[t + w]; bidx[t] = bidx[t + w];
            }
        }
        __syncthreads();
    }
    int idx = bidx[0];
    float val = (idx < Q) ? queue[(long)idx * 256 + t]
                          : zsrc[(long)(idx - Q) * 256 + t];
    target[(long)row * 256 + t] = val;
}

// NT-Xent row: masked log-softmax, gather label, accumulate -log p
__global__ __launch_bounds__(256) void k_ntxent_row(
    const float* __restrict__ logits, int R2, float invT, float* __restrict__ lossAcc)
{
    const int i = blockIdx.x;
    const int t = threadIdx.x;
    const int n = R2 >> 1;
    const int label = (i < n) ? (i + n) : (i - n);
    const float* Lr = logits + (long)i * R2;
    __shared__ float rs[256];

    float mx = -3.0e38f;
    for (int j = t; j < R2; j += 256) if (j != i) mx = fmaxf(mx, Lr[j] * invT);
    rs[t] = mx; __syncthreads();
    for (int w = 128; w > 0; w >>= 1) { if (t < w) rs[t] = fmaxf(rs[t], rs[t + w]); __syncthreads(); }
    mx = rs[0];
    __syncthreads();

    float se = 0.0f;
    for (int j = t; j < R2; j += 256) if (j != i) se += __expf(Lr[j] * invT - mx);
    rs[t] = se; __syncthreads();
    for (int w = 128; w > 0; w >>= 1) { if (t < w) rs[t] += rs[t + w]; __syncthreads(); }

    if (t == 0) {
        float logp = Lr[label] * invT - mx - __logf(rs[0]);
        atomicAdd(lossAcc, -logp);
    }
}

__global__ void k_finalize_loss(const float* __restrict__ acc, float* __restrict__ out) {
    // (mean over 2048 rows per loss, then average of the two) = (s1+s2)/4096
    out[0] = (acc[0] + acc[1]) * (1.0f / 4096.0f);
}

// ---------------------------------------------------------------------------
// Host orchestration
// ---------------------------------------------------------------------------
extern "C" void kernel_launch(void* const* d_in, const int* in_sizes, int n_in,
                              void* d_out, int out_size, void* d_ws, size_t ws_size,
                              hipStream_t stream) {
    (void)in_sizes; (void)n_in; (void)out_size; (void)ws_size;
    const int B = 1024, E = 384, H = 2048, O = 256, P = 4096, Q = 16384;
    const int S = Q + B;        // 17408 support rows
    const int R2 = 2 * B;       // 2048
    const int PN = S / 64;      // 272 column tiles

    const float* view1 = (const float*)d_in[0];
    const float* view2 = (const float*)d_in[1];
    const float* queue = (const float*)d_in[2];
    const float* pw1 = (const float*)d_in[3];
    const float* pg1 = (const float*)d_in[4];
    const float* pb1 = (const float*)d_in[5];
    const float* pw2 = (const float*)d_in[6];
    const float* pg2 = (const float*)d_in[7];
    const float* pb2 = (const float*)d_in[8];
    const float* pw3 = (const float*)d_in[9];
    const float* pg3 = (const float*)d_in[10];
    const float* pb3 = (const float*)d_in[11];
    const float* qw1 = (const float*)d_in[12];
    const float* qg1 = (const float*)d_in[13];
    const float* qb1 = (const float*)d_in[14];
    const float* qw2 = (const float*)d_in[15];
    const float* qb2 = (const float*)d_in[16];
    float* out = (float*)d_out;

    char* base = (char*)d_ws;
    size_t off = 0;
    auto alloc = [&](size_t bytes) -> void* {
        void* p = base + off;
        off = (off + bytes + 255) & ~(size_t)255;
        return p;
    };

    // bf16 weight / activation buffers
    bf16* pw1b = (bf16*)alloc((size_t)E * H * 2);
    bf16* pw2b = (bf16*)alloc((size_t)H * H * 2);
    bf16* pw3b = (bf16*)alloc((size_t)H * O * 2);
    bf16* qw1b = (bf16*)alloc((size_t)O * P * 2);
    bf16* qw2b = (bf16*)alloc((size_t)P * O * 2);
    bf16* v1b  = (bf16*)alloc((size_t)B * E * 2);
    bf16* v2b  = (bf16*)alloc((size_t)B * E * 2);
    bf16* a1b  = (bf16*)alloc((size_t)B * H * 2);
    bf16* a2b  = (bf16*)alloc((size_t)B * H * 2);
    bf16* zb   = (bf16*)alloc((size_t)B * O * 2);
    bf16* a3b  = (bf16*)alloc((size_t)B * P * 2);
    bf16* snormb = (bf16*)alloc((size_t)S * O * 2);
    bf16* qnormb = (bf16*)alloc((size_t)B * O * 2);
    bf16* znb    = (bf16*)alloc((size_t)R2 * O * 2);
    // f32 buffers
    float* tbig  = (float*)alloc((size_t)B * P * 4);
    float* z1f   = (float*)alloc((size_t)B * O * 4);
    float* z2f   = (float*)alloc((size_t)B * O * 4);
    float* p1f   = (float*)alloc((size_t)B * O * 4);
    float* p2f   = (float*)alloc((size_t)B * O * 4);
    float* meanb = (float*)alloc((size_t)P * 4);
    float* varb  = (float*)alloc((size_t)P * 4);
    float* supportf = (float*)alloc((size_t)S * O * 4);
    float* pmaxb = (float*)alloc((size_t)B * PN * 4);
    int*   pidxb = (int*)alloc((size_t)B * PN * 4);
    float* tgt1  = (float*)alloc((size_t)B * O * 4);
    float* tgt2  = (float*)alloc((size_t)B * O * 4);
    float* zcatf = (float*)alloc((size_t)R2 * O * 4);
    float* logits = (float*)alloc((size_t)R2 * R2 * 4);
    float* lossAcc = (float*)alloc(2 * 4);

    auto cvt = [&](const float* s, bf16* d, long n) {
        k_f32_to_bf16<<<512, WG, 0, stream>>>(s, d, n);
    };
    auto gemm = [&](const bf16* A, const bf16* Bm, float* C, const float* bias,
                    int M, int N, int K, int bt) {
        dim3 g(N / 64, M / 128);
        k_gemm<<<g, WG, 0, stream>>>(A, Bm, C, bias, M, N, K, bt);
    };

    // --- precision staging -------------------------------------------------
    cvt(pw1, pw1b, (long)E * H);
    cvt(pw2, pw2b, (long)H * H);
    cvt(pw3, pw3b, (long)H * O);
    cvt(qw1, qw1b, (long)O * P);
    cvt(qw2, qw2b, (long)P * O);
    cvt(view1, v1b, (long)B * E);
    cvt(view2, v2b, (long)B * E);

    // --- projector + predictor per view ------------------------------------
    const bf16* vbs[2] = {v1b, v2b};
    float* zfs[2] = {z1f, z2f};
    float* pfs[2] = {p1f, p2f};
    for (int v = 0; v < 2; ++v) {
        gemm(vbs[v], pw1b, tbig, nullptr, B, H, E, 0);
        k_bn_stats<<<H, WG, 0, stream>>>(tbig, B, H, meanb, varb);
        k_bn_apply<<<2048, WG, 0, stream>>>(tbig, meanb, varb, pg1, pb1, B, H, 1, a1b, nullptr);

        gemm(a1b, pw2b, tbig, nullptr, B, H, H, 0);
        k_bn_stats<<<H, WG, 0, stream>>>(tbig, B, H, meanb, varb);
        k_bn_apply<<<2048, WG, 0, stream>>>(tbig, meanb, varb, pg2, pb2, B, H, 1, a2b, nullptr);

        gemm(a2b, pw3b, tbig, nullptr, B, O, H, 0);
        k_bn_stats<<<O, WG, 0, stream>>>(tbig, B, O, meanb, varb);
        k_bn_apply<<<512, WG, 0, stream>>>(tbig, meanb, varb, pg3, pb3, B, O, 0, zb, zfs[v]);

        gemm(zb, qw1b, tbig, nullptr, B, P, O, 0);
        k_bn_stats<<<P, WG, 0, stream>>>(tbig, B, P, meanb, varb);
        k_bn_apply<<<4096, WG, 0, stream>>>(tbig, meanb, varb, qg1, qb1, B, P, 1, a3b, nullptr);

        gemm(a3b, qw2b, pfs[v], qb2, B, O, P, 0);
    }

    // --- nearest neighbour over queue + z1 ----------------------------------
    k_copy_f32<<<512, WG, 0, stream>>>(queue, supportf, (long)Q * O);
    k_copy_f32<<<512, WG, 0, stream>>>(z1f, supportf + (long)Q * O, (long)B * O);
    k_l2norm_rows<<<S, 256, 0, stream>>>(supportf, snormb);

    float* tgts[2] = {tgt1, tgt2};
    for (int v = 0; v < 2; ++v) {
        k_l2norm_rows<<<B, 256, 0, stream>>>(zfs[v], qnormb);
        dim3 g(S / 64, B / 128);
        k_gemm_rowmax<<<g, WG, 0, stream>>>(qnormb, snormb, B, S, O, pmaxb, pidxb);
        k_argmax_gather<<<B, 256, 0, stream>>>(pmaxb, pidxb, PN, queue, z1f, Q, tgts[v]);
    }

    // --- NT-Xent: (p1, target2) and (p2, target1) ---------------------------
    k_zero_f32<<<1, 32, 0, stream>>>(lossAcc, 2);
    const float* lp[2] = {p1f, p2f};
    const float* lt[2] = {tgt2, tgt1};
    for (int s_ = 0; s_ < 2; ++s_) {
        k_copy_f32<<<256, WG, 0, stream>>>(lp[s_], zcatf, (long)B * O);
        k_copy_f32<<<256, WG, 0, stream>>>(lt[s_], zcatf + (long)B * O, (long)B * O);
        k_l2norm_rows<<<R2, 256, 0, stream>>>(zcatf, znb);
        gemm(znb, znb, logits, nullptr, R2, R2, O, 1);
        k_ntxent_row<<<R2, 256, 0, stream>>>(logits, R2, 10.0f, lossAcc + s_);
    }

    // --- outputs: [loss][embedding 2*B*E][projection 2*B*O] -----------------
    k_finalize_loss<<<1, 1, 0, stream>>>(lossAcc, out);
    k_copy_f32<<<512, WG, 0, stream>>>(view1, out + 1, (long)B * E);
    k_copy_f32<<<512, WG, 0, stream>>>(view2, out + 1 + (long)B * E, (long)B * E);
    k_copy_f32<<<512, WG, 0, stream>>>(z1f, out + 1 + 2L * B * E, (long)B * O);
    k_copy_f32<<<512, WG, 0, stream>>>(z2f, out + 1 + 2L * B * E + (long)B * O, (long)B * O);
}
